// AttnBlock_5497558139234
// MI455X (gfx1250) — matrix operand order repaired and checked by reading, not compiled, not run on hardware
//
#include <hip/hip_runtime.h>
#include <math.h>

typedef __attribute__((ext_vector_type(16))) _Float16 v16h;
typedef __attribute__((ext_vector_type(8)))  float    v8f;

#define C_DIM 512
#define L_DIM 2048
#define B_DIM 8
#define G_DIM 32

// ------------------------------------------------------------------
// fp32 -> fp16 weight conversion
// ------------------------------------------------------------------
__global__ void k_f32_to_f16(const float* __restrict__ src,
                             _Float16* __restrict__ dst, int n) {
  int i = blockIdx.x * blockDim.x + threadIdx.x;
  if (i < n) dst[i] = (_Float16)src[i];
}

// ------------------------------------------------------------------
// GroupNorm over [B,C,L]; writes h transposed as f16: hT[b][l][c]
// One block per (batch, group): reduces 16 channels x 2048 positions.
// ------------------------------------------------------------------
__global__ __launch_bounds__(256) void k_groupnorm_t(
    const float* __restrict__ x, const float* __restrict__ gamma,
    const float* __restrict__ beta, _Float16* __restrict__ hT) {
  __shared__ float s_sum[256];
  __shared__ float s_sq[256];
  const int b = blockIdx.x >> 5;
  const int g = blockIdx.x & 31;
  const int chPerG = C_DIM / G_DIM;  // 16
  const float* xb = x + ((size_t)b * C_DIM + (size_t)g * chPerG) * L_DIM;
  const int n = chPerG * L_DIM;      // 32768

  float sum = 0.f, sq = 0.f;
  for (int idx = threadIdx.x; idx < n; idx += 256) {
    float v = xb[idx];
    sum += v;
    sq += v * v;
  }
  s_sum[threadIdx.x] = sum;
  s_sq[threadIdx.x] = sq;
  __syncthreads();
  for (int s = 128; s > 0; s >>= 1) {
    if (threadIdx.x < (unsigned)s) {
      s_sum[threadIdx.x] += s_sum[threadIdx.x + s];
      s_sq[threadIdx.x]  += s_sq[threadIdx.x + s];
    }
    __syncthreads();
  }
  const float mean = s_sum[0] / (float)n;
  const float var  = s_sq[0] / (float)n - mean * mean;
  const float rs   = rsqrtf(var + 1e-5f);

  _Float16* hTb = hT + (size_t)b * L_DIM * C_DIM;
  for (int idx = threadIdx.x; idx < n; idx += 256) {
    int c = idx >> 11;             // / 2048
    int l = idx & (L_DIM - 1);
    int ch = g * chPerG + c;
    float v = (xb[idx] - mean) * rs;
    v = v * gamma[ch] + beta[ch];
    hTb[(size_t)l * C_DIM + ch] = (_Float16)v;
  }
}

// ------------------------------------------------------------------
// Row softmax over S [nb x L_DIM x L_DIM] fp32 -> P f16.
// grid = (L_DIM, nb); one block per row.
// ------------------------------------------------------------------
__global__ __launch_bounds__(256) void k_softmax(const float* __restrict__ S,
                                                 _Float16* __restrict__ P) {
  __shared__ float red[256];
  const size_t batchOff = (size_t)blockIdx.y * L_DIM * L_DIM;
  const int row = blockIdx.x;
  const float* s = S + batchOff + (size_t)row * L_DIM;

  float vloc[8];
  float mx = -3.402823466e38f;
#pragma unroll
  for (int i = 0; i < 8; ++i) {
    float v = s[threadIdx.x + i * 256];
    vloc[i] = v;
    mx = fmaxf(mx, v);
  }
  red[threadIdx.x] = mx;
  __syncthreads();
  for (int st = 128; st > 0; st >>= 1) {
    if (threadIdx.x < (unsigned)st)
      red[threadIdx.x] = fmaxf(red[threadIdx.x], red[threadIdx.x + st]);
    __syncthreads();
  }
  mx = red[0];
  __syncthreads();

  float sum = 0.f;
#pragma unroll
  for (int i = 0; i < 8; ++i) {
    vloc[i] = __expf(vloc[i] - mx);
    sum += vloc[i];
  }
  red[threadIdx.x] = sum;
  __syncthreads();
  for (int st = 128; st > 0; st >>= 1) {
    if (threadIdx.x < (unsigned)st) red[threadIdx.x] += red[threadIdx.x + st];
    __syncthreads();
  }
  const float inv = 1.f / red[0];
  _Float16* p = P + batchOff + (size_t)row * L_DIM;
#pragma unroll
  for (int i = 0; i < 8; ++i)
    p[threadIdx.x + i * 256] = (_Float16)(vloc[i] * inv);
}

// ------------------------------------------------------------------
// WMMA GEMM:  D[M,N] = alpha * A[M,K] * B[K,N] (+bias[m]) (+resid)
//   A  : row-major [M,K] f16 (lda = K stride)
//   Bt : B stored as row-major [N,K] f16 (i.e. column-major B)
// Block tile 128x128, BK=32; 8 waves, each 32(M) x 64(N).
// Epilogue variants via template flags. Grid: (N/128, M/128, batch).
// ------------------------------------------------------------------
template <bool OUT_F16, bool TRANS_OUT, bool HAS_BIAS, bool HAS_RESID>
__global__ __launch_bounds__(256) void k_gemm_wmma(
    const _Float16* __restrict__ A, int lda, size_t strideA,
    const _Float16* __restrict__ Bt, int ldb, size_t strideB,
    void* __restrict__ outv, int ldo, size_t strideO,
    const float* __restrict__ bias,
    const float* __restrict__ resid, size_t strideR,
    int K, float alpha) {
  __shared__ _Float16 As[128][40];
  __shared__ _Float16 Bs[128][40];

  const int m0 = blockIdx.y * 128;
  const int n0 = blockIdx.x * 128;
  A  += (size_t)blockIdx.z * strideA;
  Bt += (size_t)blockIdx.z * strideB;

  const int tid  = threadIdx.x;
  const int lane = tid & 31;
  const int wave = tid >> 5;  // 0..7
  const int wm   = wave >> 1; // 0..3 (M)
  const int wn   = wave & 1;  // 0..1 (N)
  const int l15  = lane & 15;
  const int hi   = lane >> 4;

  v8f acc[2][4];
#pragma unroll
  for (int i = 0; i < 2; ++i)
#pragma unroll
    for (int j = 0; j < 4; ++j)
#pragma unroll
      for (int r = 0; r < 8; ++r) acc[i][j][r] = 0.f;

  for (int k0 = 0; k0 < K; k0 += 32) {
    __syncthreads();
    // stage 128x32 tiles of A and Bt (both row-major in K) into LDS
#pragma unroll
    for (int i = 0; i < 2; ++i) {
      int idx = tid + i * 256;       // 0..511
      int row = idx >> 2;            // 0..127
      int cg  = (idx & 3) << 3;      // 0,8,16,24
      *(uint4*)(&As[row][cg]) =
          *(const uint4*)(A + (size_t)(m0 + row) * lda + k0 + cg);
      *(uint4*)(&Bs[row][cg]) =
          *(const uint4*)(Bt + (size_t)(n0 + row) * ldb + k0 + cg);
    }
    __syncthreads();

    // A fragment: lane m=l15, K chunks at {hi*8, 16+hi*8}
    v16h af[2];
#pragma unroll
    for (int mf = 0; mf < 2; ++mf) {
      const _Float16* r = &As[wm * 32 + mf * 16 + l15][0];
      uint4* p = (uint4*)&af[mf];
      p[0] = *(const uint4*)(r + hi * 8);
      p[1] = *(const uint4*)(r + 16 + hi * 8);
    }
    // B fragment: lane n=l15, K chunks at {hi*8, 16+hi*8}
    v16h bfrag[4];
#pragma unroll
    for (int nf = 0; nf < 4; ++nf) {
      const _Float16* r = &Bs[wn * 64 + nf * 16 + l15][hi * 8];
      uint4* p = (uint4*)&bfrag[nf];
      p[0] = *(const uint4*)(r);
      p[1] = *(const uint4*)(r + 16);
    }
#pragma unroll
    for (int mf = 0; mf < 2; ++mf)
#pragma unroll
      for (int nf = 0; nf < 4; ++nf)
        acc[mf][nf] = __builtin_amdgcn_wmma_f32_16x16x32_f16(
            false, af[mf], false, bfrag[nf], (short)0, acc[mf][nf], false,
            false);
  }

  // Epilogue. D vgpr r maps to row (hi*8 + r), col l15.
#pragma unroll
  for (int mf = 0; mf < 2; ++mf) {
#pragma unroll
    for (int nf = 0; nf < 4; ++nf) {
      const int mbase = m0 + wm * 32 + mf * 16 + hi * 8;
      const int nn    = n0 + wn * 64 + nf * 16 + l15;
      float vals[8];
#pragma unroll
      for (int r = 0; r < 8; ++r) {
        float v = acc[mf][nf][r] * alpha;
        if (HAS_BIAS) v += bias[mbase + r];
        vals[r] = v;
      }
      if constexpr (TRANS_OUT) {
        _Float16* out = (_Float16*)outv + (size_t)blockIdx.z * strideO;
        alignas(16) _Float16 tmp[8];
#pragma unroll
        for (int r = 0; r < 8; ++r) tmp[r] = (_Float16)vals[r];
        *(uint4*)(&out[(size_t)nn * ldo + mbase]) = *(const uint4*)tmp;
      } else if constexpr (OUT_F16) {
        _Float16* out = (_Float16*)outv + (size_t)blockIdx.z * strideO;
#pragma unroll
        for (int r = 0; r < 8; ++r)
          out[(size_t)(mbase + r) * ldo + nn] = (_Float16)vals[r];
      } else {
        float* out = (float*)outv + (size_t)blockIdx.z * strideO;
#pragma unroll
        for (int r = 0; r < 8; ++r) {
          float v = vals[r];
          if constexpr (HAS_RESID)
            v += resid[(size_t)blockIdx.z * strideR +
                       (size_t)(mbase + r) * ldo + nn];
          out[(size_t)(mbase + r) * ldo + nn] = v;
        }
      }
    }
  }
}

// ------------------------------------------------------------------
// Host orchestration
// ------------------------------------------------------------------
extern "C" void kernel_launch(void* const* d_in, const int* in_sizes, int n_in,
                              void* d_out, int out_size, void* d_ws,
                              size_t ws_size, hipStream_t stream) {
  (void)in_sizes; (void)n_in; (void)out_size;
  const float* x     = (const float*)d_in[0];
  const float* gamma = (const float*)d_in[1];
  const float* beta  = (const float*)d_in[2];
  const float* Wq = (const float*)d_in[3];
  const float* bq = (const float*)d_in[4];
  const float* Wk = (const float*)d_in[5];
  const float* bk = (const float*)d_in[6];
  const float* Wv = (const float*)d_in[7];
  const float* bv = (const float*)d_in[8];
  const float* Wo = (const float*)d_in[9];
  const float* bo = (const float*)d_in[10];
  float* out = (float*)d_out;

  char* ws = (char*)d_ws;
  size_t off = 0;
  const size_t szW  = (size_t)C_DIM * C_DIM;          // 262144 elems
  const size_t szLC = (size_t)L_DIM * C_DIM;          // 1M elems per batch
  const size_t szH  = (size_t)B_DIM * szLC;           // 8M elems
  const size_t szLL = (size_t)L_DIM * L_DIM;          // 4M elems per batch

  _Float16* wq16 = (_Float16*)(ws + off); off += szW * 2;
  _Float16* wk16 = (_Float16*)(ws + off); off += szW * 2;
  _Float16* wv16 = (_Float16*)(ws + off); off += szW * 2;
  _Float16* wo16 = (_Float16*)(ws + off); off += szW * 2;
  _Float16* hT   = (_Float16*)(ws + off); off += szH * 2;  // [B,L,C]
  _Float16* qT   = (_Float16*)(ws + off); off += szH * 2;  // [B,L,C]
  _Float16* kT   = (_Float16*)(ws + off); off += szH * 2;  // [B,L,C]
  _Float16* vbuf = (_Float16*)(ws + off); off += szH * 2;  // [B,C,L]
  _Float16* aT   = (_Float16*)(ws + off); off += szH * 2;  // [B,L,C]

  // How many batches' worth of attention scratch (S fp32 + P f16) fit?
  const size_t perSP = szLL * 6;  // 4B S + 2B P per element
  int slots = 1;
  if (ws_size > off + perSP) {
    size_t s = (ws_size - off) / perSP;
    slots = (s >= (size_t)B_DIM) ? B_DIM : (int)s;
    if (slots < 1) slots = 1;
  }
  float*    S = (float*)(ws + off);
  _Float16* P = (_Float16*)(ws + off + (size_t)slots * szLL * 4);

  // 1) weights fp32 -> fp16
  int nW = (int)szW;
  k_f32_to_f16<<<(nW + 255) / 256, 256, 0, stream>>>(Wq, wq16, nW);
  k_f32_to_f16<<<(nW + 255) / 256, 256, 0, stream>>>(Wk, wk16, nW);
  k_f32_to_f16<<<(nW + 255) / 256, 256, 0, stream>>>(Wv, wv16, nW);
  k_f32_to_f16<<<(nW + 255) / 256, 256, 0, stream>>>(Wo, wo16, nW);

  // 2) GroupNorm -> hT f16
  k_groupnorm_t<<<B_DIM * G_DIM, 256, 0, stream>>>(x, gamma, beta, hT);

  // 3) q/k/v projections (batched over grid.z)
  dim3 gProj(L_DIM / 128, C_DIM / 128, B_DIM);
  k_gemm_wmma<true, true, true, false><<<gProj, 256, 0, stream>>>(
      wq16, C_DIM, 0, hT, C_DIM, szLC, (void*)qT, C_DIM, szLC, bq, nullptr, 0,
      C_DIM, 1.0f);
  k_gemm_wmma<true, true, true, false><<<gProj, 256, 0, stream>>>(
      wk16, C_DIM, 0, hT, C_DIM, szLC, (void*)kT, C_DIM, szLC, bk, nullptr, 0,
      C_DIM, 1.0f);
  k_gemm_wmma<true, false, true, false><<<gProj, 256, 0, stream>>>(
      wv16, C_DIM, 0, hT, C_DIM, szLC, (void*)vbuf, L_DIM, szLC, bv, nullptr,
      0, C_DIM, 1.0f);

  // 4) attention, processed in chunks of `slots` batches
  const float scale = 1.0f / sqrtf((float)C_DIM);
  for (int b0 = 0; b0 < B_DIM; b0 += slots) {
    int nb = B_DIM - b0;
    if (nb > slots) nb = slots;
    dim3 gS(L_DIM / 128, L_DIM / 128, nb);
    dim3 gSm(L_DIM, nb, 1);
    dim3 gPV(L_DIM / 128, C_DIM / 128, nb);
    // S[i,j] = scale * sum_c q[c,i] k[c,j]
    k_gemm_wmma<false, false, false, false><<<gS, 256, 0, stream>>>(
        qT + (size_t)b0 * szLC, C_DIM, szLC, kT + (size_t)b0 * szLC, C_DIM,
        szLC, (void*)S, L_DIM, szLL, nullptr, nullptr, 0, C_DIM, scale);
    k_softmax<<<gSm, 256, 0, stream>>>(S, P);
    // a[c,i] = sum_j v[c,j] P[i,j]  -> stored transposed aT[i,c]
    k_gemm_wmma<true, true, false, false><<<gPV, 256, 0, stream>>>(
        vbuf + (size_t)b0 * szLC, L_DIM, szLC, P, L_DIM, szLL,
        (void*)(aT + (size_t)b0 * szLC), C_DIM, szLC, nullptr, nullptr, 0,
        L_DIM, 1.0f);
  }

  // 5) output projection + bias + residual
  k_gemm_wmma<false, false, true, true><<<gProj, 256, 0, stream>>>(
      wo16, C_DIM, 0, aT, C_DIM, szLC, (void*)out, L_DIM, szLC, bo, x, szLC,
      C_DIM, 1.0f);
}
